// DecomposingAttnProcessor_41669772706194
// MI455X (gfx1250) — matrix operand order repaired and checked by reading, not compiled, not run on hardware
//
#include <hip/hip_runtime.h>

typedef __bf16 bf16_t;
typedef __bf16 v4bf  __attribute__((ext_vector_type(4)));
typedef __bf16 v8bf  __attribute__((ext_vector_type(8)));
typedef __bf16 v16bf __attribute__((ext_vector_type(16)));
typedef float  v8f   __attribute__((ext_vector_type(8)));

// Problem constants (from reference setup_inputs)
constexpr int kBC = 8;      // batch*components
constexpr int kS  = 4096;   // spatial tokens
constexpr int kD  = 1536;   // hidden dim
constexpr int kT  = 154;    // text tokens
constexpr int kH  = 24;     // heads
constexpr int kDH = 64;     // dim per head
constexpr int kC  = 4;      // components
constexpr int kB  = 2;      // batch = BC / C
constexpr int kTP = 160;    // T padded to multiple of 32 (5 K-steps of 32)
constexpr float kEPS = 1e-8f;

__device__ inline v8f wmma_bf16(v16bf a, v16bf b, v8f c) {
  // v_wmma_f32_16x16x32_bf16: D = A(16x32) * B(32x16) + C, f32 accumulate
  return __builtin_amdgcn_wmma_f32_16x16x32_bf16(
      /*neg_a=*/false, a, /*neg_b=*/false, b,
      /*c_mod=*/(short)0, c, /*reuse_a=*/false, /*reuse_b=*/false);
}

// ---------------------------------------------------------------------------
// Kernel 1: f32 -> bf16 conversion (vectorized x4; all sizes divisible by 4)
// ---------------------------------------------------------------------------
__global__ void cvt_f32_to_bf16(const float* __restrict__ src,
                                bf16_t* __restrict__ dst, size_t n4) {
  size_t i = (size_t)blockIdx.x * blockDim.x + threadIdx.x;
  size_t stride = (size_t)gridDim.x * blockDim.x;
  const float4* s4 = (const float4*)src;
  v4bf* d4 = (v4bf*)dst;
  for (; i < n4; i += stride) {
    float4 f = s4[i];
    v4bf o;
    o[0] = (bf16_t)f.x; o[1] = (bf16_t)f.y;
    o[2] = (bf16_t)f.z; o[3] = (bf16_t)f.w;
    d4[i] = o;
  }
}

// ---------------------------------------------------------------------------
// Kernel 2: C[M,N] = A[M,K] @ W[N,K]^T + bias  (torch Linear semantics)
// A, W in bf16; f32 accumulate; output bf16 or f32.
//
// Block = 256 thr = 8 waves. Wave register tile: 32(M) x 64(N) =
// 2 A-frags x 4 B-frags -> 8 accumulators (64 VGPRs). Per K-step a wave
// loads 2 KB A + 4 KB B for 8 WMMAs (~22 FLOP/B from cache); A (<=100MB)
// and all weights (19MB) stay L2-resident.
//
// Fragment layouts used here (contiguous k order, not the ISA table order; wave32):
//  A 16x32: lane(m=lane&15, half=lane>>4); regs j = K pair (16*half+2j,+2j+1)
//           -> one contiguous 32B chunk of row m (two 16B loads).
//  B 32x16: lane(n=lane&15, half);  regs j = K pair (16*half+2j,+2j+1)
//           -> one contiguous 32B chunk of row n of W.
//  C/D 16x16 f32: lane(n, half); reg j -> row m = j + 8*half, col n.
// ---------------------------------------------------------------------------
__device__ inline v16bf load_afrag(const bf16_t* a_row, int k0, int half) {
  v8bf a_lo = *(const v8bf*)(a_row + k0 + 16 * half);
  v8bf a_hi = *(const v8bf*)(a_row + k0 + 16 * half + 8);
  return __builtin_shufflevector(a_lo, a_hi,
      0,1,2,3,4,5,6,7,8,9,10,11,12,13,14,15);
}

template <int MR, bool OUT_F32>
__device__ void gemm_tile(const bf16_t* __restrict__ A,
                          const bf16_t* __restrict__ W,
                          const float* __restrict__ bias,
                          void* __restrict__ Cout,
                          int m_base, int n_base, int N, int K,
                          int half, int lm) {
  v8f acc[MR][4] = {};
  const bf16_t* a_row[MR];
#pragma unroll
  for (int mr = 0; mr < MR; ++mr)
    a_row[mr] = A + (size_t)(m_base + mr * 16 + lm) * K;

  for (int k0 = 0; k0 < K; k0 += 32) {
    v16bf afrag[MR];
#pragma unroll
    for (int mr = 0; mr < MR; ++mr) afrag[mr] = load_afrag(a_row[mr], k0, half);
#pragma unroll
    for (int nf = 0; nf < 4; ++nf) {
      const bf16_t* w_ptr =
          W + (size_t)(n_base + nf * 16 + lm) * K + k0 + 16 * half;
      v16bf bfrag = *(const v16bf*)w_ptr;   // L2-resident weights
#pragma unroll
      for (int mr = 0; mr < MR; ++mr)
        acc[mr][nf] = wmma_bf16(afrag[mr], bfrag, acc[mr][nf]);
    }
  }

#pragma unroll
  for (int nf = 0; nf < 4; ++nf) {
    const int n = n_base + nf * 16 + lm;
    const float bv = bias[n];
#pragma unroll
    for (int mr = 0; mr < MR; ++mr) {
#pragma unroll
      for (int j = 0; j < 8; ++j) {
        const int m = m_base + mr * 16 + j + 8 * half;
        float val = acc[mr][nf][j] + bv;
        if (OUT_F32) ((float*)Cout)[(size_t)m * N + n] = val;
        else         ((bf16_t*)Cout)[(size_t)m * N + n] = (bf16_t)val;
      }
    }
  }
}

template <bool OUT_F32>
__global__ void __launch_bounds__(256)
gemm_xwT(const bf16_t* __restrict__ A, const bf16_t* __restrict__ W,
         const float* __restrict__ bias, void* __restrict__ Cout,
         int M, int N, int K) {
  const int lane = threadIdx.x & 31;
  const int wave = threadIdx.x >> 5;
  const int half = lane >> 4;
  const int lm   = lane & 15;
  const int m_base = blockIdx.y * 256 + wave * 32;
  if (m_base >= M) return;                 // wave-uniform: EXEC stays all-1s
  const int n_base = blockIdx.x * 64;

  if (m_base + 16 < M) {                   // wave-uniform branch
    gemm_tile<2, OUT_F32>(A, W, bias, Cout, m_base, n_base, N, K, half, lm);
  } else {
    gemm_tile<1, OUT_F32>(A, W, bias, Cout, m_base, n_base, N, K, half, lm);
  }
}

// ---------------------------------------------------------------------------
// Kernel 3: fused decomposing attention.
// grid = (S/16, H, B); block = 128 thr = 4 waves, wave c = component c.
//   bc = c*B + b   (reference: scores.reshape(C, B, H, S, T))
// Per block: scores[4][16][TP] (f32, LDS) via WMMA; softmax across the 4
// components per (s,t); per-row 1/(sum_t w + eps); then w @ V via WMMA with
// V^T staged in LDS as bf16.
// ---------------------------------------------------------------------------
__global__ void __launch_bounds__(128)
attn_kernel(const bf16_t* __restrict__ qp, const bf16_t* __restrict__ kp,
            const bf16_t* __restrict__ vp, bf16_t* __restrict__ op) {
  __shared__ __align__(32) float  sc[kC][16][kTP];   // 40 KB scores/weights
  __shared__              float  rs[kC][16];         // 1/(rowsum+eps)
  __shared__ __align__(32) bf16_t vt[kC][kDH][kTP];  // 80 KB V^T per comp

  const int tid  = threadIdx.x;
  const int lane = tid & 31;
  const int c    = tid >> 5;          // component
  const int half = lane >> 4;
  const int lm   = lane & 15;
  const int s_base = blockIdx.x * 16;
  const int h      = blockIdx.y;
  const int b      = blockIdx.z;
  const int bc     = c * kB + b;

  // ---- stage V^T into LDS (zero-pad t >= T) ----
  {
    const bf16_t* vbase = vp + (size_t)bc * kT * kD + h * kDH;
    for (int t = lane; t < kTP; t += 32) {
      if (t < kT) {
        const bf16_t* vrow = vbase + (size_t)t * kD;
#pragma unroll
        for (int i = 0; i < kDH / 8; ++i) {
          v8bf chunk = *(const v8bf*)(vrow + i * 8);
#pragma unroll
          for (int j = 0; j < 8; ++j) vt[c][i * 8 + j][t] = chunk[j];
        }
      } else {
#pragma unroll
        for (int n = 0; n < kDH; ++n) vt[c][n][t] = (bf16_t)0.0f;
      }
    }
  }

  // ---- scores = scale * Q K^T  (16 x TP per component) ----
  const float scale = 0.125f;  // 64^-0.5
  const bf16_t* a_row = qp + ((size_t)bc * kS + s_base + lm) * kD + h * kDH;
#pragma unroll
  for (int tt = 0; tt < kTP / 16; ++tt) {
    v8f acc = {};
    const int t = tt * 16 + lm;  // B-frag column -> key index
#pragma unroll
    for (int ks = 0; ks < kDH; ks += 32) {
      v16bf afrag = load_afrag(a_row, ks, half);
      v16bf bfrag = {};
      if (t < kT) {
        bfrag = *(const v16bf*)(kp + ((size_t)bc * kT + t) * kD +
                                h * kDH + ks + 16 * half);
      }
      acc = wmma_bf16(afrag, bfrag, acc);
    }
#pragma unroll
    for (int j = 0; j < 8; ++j)
      sc[c][j + 8 * half][tt * 16 + lm] = acc[j] * scale;
  }
  __syncthreads();

  // ---- softmax across the 4 components at each (s, t); mask t >= T ----
  for (int idx = tid; idx < 16 * kTP; idx += 128) {
    const int srow = idx / kTP;
    const int t    = idx % kTP;
    if (t < kT) {
      float x0 = sc[0][srow][t], x1 = sc[1][srow][t];
      float x2 = sc[2][srow][t], x3 = sc[3][srow][t];
      float mx = fmaxf(fmaxf(x0, x1), fmaxf(x2, x3));
      float e0 = __expf(x0 - mx), e1 = __expf(x1 - mx);
      float e2 = __expf(x2 - mx), e3 = __expf(x3 - mx);
      float inv = 1.0f / (e0 + e1 + e2 + e3);
      sc[0][srow][t] = e0 * inv; sc[1][srow][t] = e1 * inv;
      sc[2][srow][t] = e2 * inv; sc[3][srow][t] = e3 * inv;
    } else {
      sc[0][srow][t] = 0.0f; sc[1][srow][t] = 0.0f;
      sc[2][srow][t] = 0.0f; sc[3][srow][t] = 0.0f;
    }
  }
  __syncthreads();

  // ---- per-(component,row) key normalization factor ----
  if (tid < 64) {
    const int cc = tid >> 4, srow = tid & 15;
    float sum = 0.0f;
    for (int t = 0; t < kT; ++t) sum += sc[cc][srow][t];
    rs[cc][srow] = 1.0f / (sum + kEPS);
  }
  __syncthreads();

  // ---- out = (w * rowscale) @ V  : 16 x 64 per component ----
  v8f acc2[4] = {};
  const float wscale = rs[c][lm];   // A-frag row is lm for every lane
  for (int k0 = 0; k0 < kTP; k0 += 32) {
    v16bf afrag;
#pragma unroll
    for (int i = 0; i < 8; ++i) {
      afrag[i]     = (bf16_t)(sc[c][lm][k0 + 16 * half + i] * wscale);
      afrag[8 + i] = (bf16_t)(sc[c][lm][k0 + 16 * half + 8 + i] * wscale);
    }
#pragma unroll
    for (int nf = 0; nf < 4; ++nf) {
      v16bf bfrag = *(const v16bf*)(&vt[c][nf * 16 + lm][k0 + 16 * half]);
      acc2[nf] = wmma_bf16(afrag, bfrag, acc2[nf]);
    }
  }
#pragma unroll
  for (int nf = 0; nf < 4; ++nf) {
#pragma unroll
    for (int j = 0; j < 8; ++j) {
      op[((size_t)bc * kS + s_base + j + 8 * half) * kD +
         h * kDH + nf * 16 + lm] = (bf16_t)acc2[nf][j];
    }
  }
}

// ---------------------------------------------------------------------------
// Host orchestration
// ---------------------------------------------------------------------------
extern "C" void kernel_launch(void* const* d_in, const int* in_sizes, int n_in,
                              void* d_out, int out_size, void* d_ws,
                              size_t ws_size, hipStream_t stream) {
  const float* hs  = (const float*)d_in[0];
  const float* ehs = (const float*)d_in[1];
  const float* Wq  = (const float*)d_in[2];
  const float* bq  = (const float*)d_in[3];
  const float* Wk  = (const float*)d_in[4];
  const float* bk  = (const float*)d_in[5];
  const float* Wv  = (const float*)d_in[6];
  const float* bv  = (const float*)d_in[7];
  const float* Wo  = (const float*)d_in[8];
  const float* bo  = (const float*)d_in[9];

  char* ws = (char*)d_ws;
  auto alloc = [&](size_t bytes) -> char* {
    char* p = ws;
    ws += (bytes + 255) & ~(size_t)255;
    return p;
  };

  const size_t nHS  = (size_t)kBC * kS * kD;  // 50,331,648
  const size_t nEHS = (size_t)kBC * kT * kD;  //  1,892,352
  const size_t nW   = (size_t)kD * kD;        //  2,359,296

  bf16_t* hs_b   = (bf16_t*)alloc(nHS * 2);
  bf16_t* ehs_b  = (bf16_t*)alloc(nEHS * 2);
  bf16_t* Wq_b   = (bf16_t*)alloc(nW * 2);
  bf16_t* Wk_b   = (bf16_t*)alloc(nW * 2);
  bf16_t* Wv_b   = (bf16_t*)alloc(nW * 2);
  bf16_t* Wo_b   = (bf16_t*)alloc(nW * 2);
  bf16_t* q_b    = (bf16_t*)alloc(nHS * 2);
  bf16_t* k_b    = (bf16_t*)alloc(nEHS * 2);
  bf16_t* v_b    = (bf16_t*)alloc(nEHS * 2);
  bf16_t* attn_b = (bf16_t*)alloc(nHS * 2);

  auto cvt = [&](const float* s, bf16_t* d, size_t n) {
    size_t n4 = n / 4;
    int blocks = (int)((n4 + 1023) / 1024);
    if (blocks < 1) blocks = 1;
    cvt_f32_to_bf16<<<blocks, 256, 0, stream>>>(s, d, n4);
  };
  cvt(hs,  hs_b,  nHS);
  cvt(ehs, ehs_b, nEHS);
  cvt(Wq, Wq_b, nW);
  cvt(Wk, Wk_b, nW);
  cvt(Wv, Wv_b, nW);
  cvt(Wo, Wo_b, nW);

  const int Mq = kBC * kS;   // 32768
  const int Mk = kBC * kT;   // 1232 (divisible by 16)

  // Q / K / V projections (bf16 out)
  gemm_xwT<false><<<dim3(kD / 64, Mq / 256), 256, 0, stream>>>(
      hs_b, Wq_b, bq, q_b, Mq, kD, kD);
  gemm_xwT<false><<<dim3(kD / 64, (Mk + 255) / 256), 256, 0, stream>>>(
      ehs_b, Wk_b, bk, k_b, Mk, kD, kD);
  gemm_xwT<false><<<dim3(kD / 64, (Mk + 255) / 256), 256, 0, stream>>>(
      ehs_b, Wv_b, bv, v_b, Mk, kD, kD);

  // Fused decomposing attention
  attn_kernel<<<dim3(kS / 16, kH, kB), 128, 0, stream>>>(q_b, k_b, v_b, attn_b);

  // Output projection (f32 out -> d_out)
  gemm_xwT<true><<<dim3(kD / 64, Mq / 256), 256, 0, stream>>>(
      attn_b, Wo_b, bo, (float*)d_out, Mq, kD, kD);
}
